// MultiHeadAttention_33320356282741
// MI455X (gfx1250) — matrix operand order repaired and checked by reading, not compiled, not run on hardware
//
#include <hip/hip_runtime.h>

typedef __attribute__((ext_vector_type(16))) __bf16 v16bf;
typedef __attribute__((ext_vector_type(2)))  __bf16 v2bf;
typedef __attribute__((ext_vector_type(8)))  float  v8f;

#define D_MODEL 1024
#define NHEADS  16
#define DK      64
#define SEQ     2048

// --------------------------------------------------------------------------
// WMMA helper: D = A(16x32 bf16) x B(32x16 bf16) + C(16x16 f32)
// --------------------------------------------------------------------------
__device__ __forceinline__ v8f wmma_bf16(v16bf a, v16bf b, v8f c) {
  return __builtin_amdgcn_wmma_f32_16x16x32_bf16(
      /*neg_a=*/false, a, /*neg_b=*/false, b,
      /*c_mod=*/(short)0, c, /*reuse_a=*/false, /*reuse_b=*/false);
}

// A-frag k order used here (B frags match): lanes 0-15 hold K=i,
// lanes 16-31 hold K=16+i.  One contiguous 16-element run -> 2x b128 loads.
__device__ __forceinline__ int afrag_k(int lane, int i) {
  return i + ((lane >> 4) << 4);
}

// Packed f32x2 -> bf16x2 (lets the backend pick v_cvt_pk_bf16_f32)
__device__ __forceinline__ v2bf pk_bf16(float lo, float hi) {
  v2bf r; r.x = (__bf16)lo; r.y = (__bf16)hi; return r;
}

// Async global->LDS 16-byte copy (CDNA5 GLOBAL_LOAD_ASYNC_TO_LDS_B128).
// ldsOff = low 32 bits of generic shared pointer == wave-relative LDS address.
__device__ __forceinline__ void async_copy_b128(unsigned ldsOff, const void* g) {
  asm volatile("global_load_async_to_lds_b128 %0, %1, off"
               :: "v"(ldsOff), "v"(g) : "memory");
}
__device__ __forceinline__ void async_wait0() {
  asm volatile("s_wait_asynccnt 0x0" ::: "memory");
}

// ---- staging helpers (tile shapes fixed: A 128x32, Bt 128x32 transposed) ----
__device__ __forceinline__ void stage_f32_A(__bf16 (*sA)[32], const float* xbase,
                                            int tid) {
  const int idx = tid * 16;
  const int row = idx >> 5, col = idx & 31;   // col in {0,16}
  const float* src = xbase + (size_t)row * D_MODEL + col;
  float4 f0 = *(const float4*)(src);
  float4 f1 = *(const float4*)(src + 4);
  float4 f2 = *(const float4*)(src + 8);
  float4 f3 = *(const float4*)(src + 12);
  v2bf* d = (v2bf*)&sA[row][col];
  d[0] = pk_bf16(f0.x, f0.y);  d[1] = pk_bf16(f0.z, f0.w);
  d[2] = pk_bf16(f1.x, f1.y);  d[3] = pk_bf16(f1.z, f1.w);
  d[4] = pk_bf16(f2.x, f2.y);  d[5] = pk_bf16(f2.z, f2.w);
  d[6] = pk_bf16(f3.x, f3.y);  d[7] = pk_bf16(f3.z, f3.w);
}

__device__ __forceinline__ void stage_f32_Bt(__bf16 (*sBt)[32], const float* Wbase,
                                             int tid) {
  const int pidx = tid * 8;            // 8 columns / thread
  const int kp   = pidx >> 7;          // k pair 0..15 -> k = 2*kp
  const int col  = pidx & 127;
  const float* r0 = Wbase + (size_t)(2 * kp) * D_MODEL + col;
  const float* r1 = r0 + D_MODEL;
  float4 a0 = *(const float4*)(r0), a1 = *(const float4*)(r0 + 4);
  float4 b0 = *(const float4*)(r1), b1 = *(const float4*)(r1 + 4);
  const float ra[8] = {a0.x,a0.y,a0.z,a0.w,a1.x,a1.y,a1.z,a1.w};
  const float rb[8] = {b0.x,b0.y,b0.z,b0.w,b1.x,b1.y,b1.z,b1.w};
  #pragma unroll
  for (int j = 0; j < 8; j++)
    *(v2bf*)&sBt[col + j][2 * kp] = pk_bf16(ra[j], rb[j]);
}

// ==========================================================================
// Kernel 1: fused QKV projection. Y = x @ W{q,k,v} + b (f32 in, bf16 out).
// Outputs stored [b, h, s, d].  Double-buffered LDS pipeline, one barrier
// per K-step: stage(next); wmma(cur); barrier.
// ==========================================================================
__global__ __launch_bounds__(256) void qkv_gemm_kernel(
    const float* __restrict__ x,
    const float* __restrict__ Wq, const float* __restrict__ bq,
    const float* __restrict__ Wk, const float* __restrict__ bk,
    const float* __restrict__ Wv, const float* __restrict__ bv,
    __bf16* __restrict__ Qo, __bf16* __restrict__ Ko, __bf16* __restrict__ Vo)
{
  __shared__ __bf16 sA [2][128][32];   // [buf][row][k]
  __shared__ __bf16 sBt[2][128][32];   // [buf][col][k]

  const int nb = blockIdx.x * 128;
  const int mb = blockIdx.y * 128;
  const int z  = blockIdx.z;
  const float* W    = (z == 0) ? Wq : (z == 1) ? Wk : Wv;
  const float* bias = (z == 0) ? bq : (z == 1) ? bk : bv;

  const int tid  = threadIdx.x;
  const int wave = tid >> 5, lane = tid & 31;
  const int lh   = lane >> 4, ln  = lane & 15;
  const int wm   = (wave >> 1) * 32;
  const int wn   = (wave & 1)  * 64;

  const float* xb = &x[(size_t)mb * D_MODEL];
  const float* wb = &W[nb];

  v8f zero = {};
  v8f acc[2][4];
  #pragma unroll
  for (int mt = 0; mt < 2; mt++)
    #pragma unroll
    for (int nt = 0; nt < 4; nt++) acc[mt][nt] = zero;

  // prologue: stage tile 0
  stage_f32_A (sA [0], xb, tid);
  stage_f32_Bt(sBt[0], wb, tid);
  __syncthreads();

  for (int k0 = 0; k0 < D_MODEL; k0 += 32) {
    const int cur = (k0 >> 5) & 1;
    if (k0 + 32 < D_MODEL) {   // prefetch next tile into other buffer
      stage_f32_A (sA [cur ^ 1], xb + (k0 + 32), tid);
      stage_f32_Bt(sBt[cur ^ 1], wb + (size_t)(k0 + 32) * D_MODEL, tid);
    }

    v16bf aF[2], bF[4];
    #pragma unroll
    for (int mt = 0; mt < 2; mt++)
      #pragma unroll
      for (int i = 0; i < 16; i++)
        aF[mt][i] = sA[cur][wm + mt * 16 + ln][afrag_k(lane, i)];
    #pragma unroll
    for (int nt = 0; nt < 4; nt++)
      #pragma unroll
      for (int i = 0; i < 16; i++)        // contiguous: 2x ds_load_b128
        bF[nt][i] = sBt[cur][wn + nt * 16 + ln][lh * 16 + i];

    #pragma unroll
    for (int mt = 0; mt < 2; mt++)
      #pragma unroll
      for (int nt = 0; nt < 4; nt++)
        acc[mt][nt] = wmma_bf16(aF[mt], bF[nt], acc[mt][nt]);
    __syncthreads();
  }

  // ---- epilogue: bias, convert, scatter to [b,h,s,d]
  __bf16* dst = (z == 0) ? Qo : (z == 1) ? Ko : Vo;
  #pragma unroll
  for (int mt = 0; mt < 2; mt++)
    #pragma unroll
    for (int nt = 0; nt < 4; nt++)
      #pragma unroll
      for (int r = 0; r < 8; r++) {
        const int grow = mb + wm + mt * 16 + r + lh * 8;   // b*S + s
        const int gcol = nb + wn + nt * 16 + ln;           // h*64 + d
        const float vv = acc[mt][nt][r] + bias[gcol];
        const int b = grow >> 11, s = grow & (SEQ - 1);
        const int h = gcol >> 6,  d = gcol & (DK - 1);
        dst[((size_t)(b * NHEADS + h) * SEQ + s) * DK + d] = (__bf16)vv;
      }
}

// ==========================================================================
// Kernel 2: flash attention, double-buffered K/V tiles.  K tile via async
// global->LDS copies; V tile transposed on stage so P@V B-frags are
// contiguous b128 reads.  One barrier per 32-key step.
// ==========================================================================
__device__ __forceinline__ void stage_kv(__bf16 (*sK)[DK], __bf16 (*sVt)[32],
                                         const __bf16* Kh, const __bf16* Vh,
                                         int kb, int tid) {
  { // K tile (32 keys x 64 d): 16B async copy per thread
    const int idx = tid * 8;
    const int key = idx >> 6, d0 = idx & 63;
    async_copy_b128((unsigned)(size_t)&sK[key][d0],
                    Kh + (size_t)(kb + key) * DK + d0);
  }
  { // V tile transposed: two keys x 4 d per thread, paired b32 stores
    const int kp = tid >> 4;            // key pair 0..15
    const int d0 = (tid & 15) * 4;
    ushort4 a = *(const ushort4*)(Vh + (size_t)(kb + 2 * kp)     * DK + d0);
    ushort4 b = *(const ushort4*)(Vh + (size_t)(kb + 2 * kp + 1) * DK + d0);
    *(unsigned*)&sVt[d0 + 0][2 * kp] = (unsigned)a.x | ((unsigned)b.x << 16);
    *(unsigned*)&sVt[d0 + 1][2 * kp] = (unsigned)a.y | ((unsigned)b.y << 16);
    *(unsigned*)&sVt[d0 + 2][2 * kp] = (unsigned)a.z | ((unsigned)b.z << 16);
    *(unsigned*)&sVt[d0 + 3][2 * kp] = (unsigned)a.w | ((unsigned)b.w << 16);
  }
}

__global__ __launch_bounds__(256) void attn_kernel(
    const __bf16* __restrict__ Q, const __bf16* __restrict__ K,
    const __bf16* __restrict__ V, __bf16* __restrict__ AO)
{
  __shared__ __bf16 sK [2][32][DK];    // [buf][key][d]   (async loaded)
  __shared__ __bf16 sVt[2][DK][32];    // [buf][d][key]   (transposed)
  __shared__ __bf16 sP [8][16][32];    // per-wave P relayout bounce

  const int bh   = blockIdx.y;                 // b*16 + h
  const int tid  = threadIdx.x;
  const int wave = tid >> 5, lane = tid & 31;
  const int lh   = lane >> 4, ln  = lane & 15;
  const int qRow = blockIdx.x * 128 + wave * 16;

  const __bf16* Qh = Q + (size_t)bh * SEQ * DK;
  const __bf16* Kh = K + (size_t)bh * SEQ * DK;
  const __bf16* Vh = V + (size_t)bh * SEQ * DK;

  // Q fragments (16x64 = two 16x32 A-frags), 1/sqrt(Dk) folded in
  v16bf qa[2];
  #pragma unroll
  for (int c = 0; c < 2; c++)
    #pragma unroll
    for (int i = 0; i < 16; i++) {
      const int d = c * 32 + afrag_k(lane, i);
      const float qv = (float)Qh[(size_t)(qRow + ln) * DK + d];
      qa[c][i] = (__bf16)(qv * 0.125f);
    }

  v8f zero = {};
  v8f o[4];
  #pragma unroll
  for (int t = 0; t < 4; t++) o[t] = zero;
  float m[8], lsum[8];
  #pragma unroll
  for (int r = 0; r < 8; r++) { m[r] = -3.0e38f; lsum[r] = 0.0f; }

  // prologue: stage tile 0
  stage_kv(sK[0], sVt[0], Kh, Vh, 0, tid);
  async_wait0();
  __syncthreads();

  for (int kb = 0; kb < SEQ; kb += 32) {
    const int cur = (kb >> 5) & 1;
    if (kb + 32 < SEQ)       // prefetch next tile into other buffer
      stage_kv(sK[cur ^ 1], sVt[cur ^ 1], Kh, Vh, kb + 32, tid);

    // ---- scores: 16 q-rows x 32 keys; B-frag = contiguous d run in sK
    v8f sc0 = zero, sc1 = zero;
    #pragma unroll
    for (int c = 0; c < 2; c++) {
      v16bf kf0, kf1;
      #pragma unroll
      for (int i = 0; i < 16; i++) {
        const int d = c * 32 + lh * 16 + i;
        kf0[i] = sK[cur][ln][d];
        kf1[i] = sK[cur][16 + ln][d];
      }
      sc0 = wmma_bf16(qa[c], kf0, sc0);
      sc1 = wmma_bf16(qa[c], kf1, sc1);
    }

    // ---- online softmax
    #pragma unroll
    for (int r = 0; r < 8; r++) {
      const float s0 = sc0[r], s1 = sc1[r];
      float mx = fmaxf(s0, s1);
      #pragma unroll
      for (int off = 1; off < 16; off <<= 1)
        mx = fmaxf(mx, __shfl_xor(mx, off, 32));
      const float mnew  = fmaxf(m[r], mx);
      const float alpha = __expf(m[r] - mnew);
      const float p0 = __expf(s0 - mnew);
      const float p1 = __expf(s1 - mnew);
      float rs = p0 + p1;
      #pragma unroll
      for (int off = 1; off < 16; off <<= 1)
        rs += __shfl_xor(rs, off, 32);
      lsum[r] = lsum[r] * alpha + rs;
      m[r] = mnew;
      #pragma unroll
      for (int t = 0; t < 4; t++) o[t][r] *= alpha;
      const int prow = r + lh * 8;
      sP[wave][prow][ln]      = (__bf16)p0;
      sP[wave][prow][16 + ln] = (__bf16)p1;
    }

    // ---- P (16x32) as A-frag via LDS bounce (same-wave DS ops in-order)
    v16bf pa;
    #pragma unroll
    for (int i = 0; i < 16; i++)
      pa[i] = sP[wave][ln][afrag_k(lane, i)];

    // ---- O += P @ V; B-frag = contiguous key run in sVt
    #pragma unroll
    for (int t = 0; t < 4; t++) {
      v16bf vf;
      #pragma unroll
      for (int i = 0; i < 16; i++)
        vf[i] = sVt[cur][t * 16 + ln][lh * 16 + i];
      o[t] = wmma_bf16(pa, vf, o[t]);
    }

    async_wait0();      // prefetch for next buffer must have landed
    __syncthreads();
  }

  // ---- normalize, write AO [b, s, h*64+d]
  const int b = bh >> 4, h = bh & (NHEADS - 1);
  #pragma unroll
  for (int t = 0; t < 4; t++)
    #pragma unroll
    for (int r = 0; r < 8; r++) {
      const int s = qRow + r + lh * 8;
      const int d = t * 16 + ln;
      const float val = o[t][r] / lsum[r];
      AO[((size_t)b * SEQ + s) * D_MODEL + h * DK + d] = (__bf16)val;
    }
}

// ==========================================================================
// Kernel 3: output projection. out = AO(bf16) @ Wo(f32->bf16) + bo, f32 out.
// A tile staged via async global->LDS copies; double-buffered pipeline.
// ==========================================================================
__device__ __forceinline__ void stage_bf16_A_async(__bf16 (*sA)[32],
                                                   const __bf16* Abase, int tid) {
  const int idx = tid * 16;
  const int row = idx >> 5, col = idx & 31;
  const __bf16* src = Abase + (size_t)row * D_MODEL + col;
  async_copy_b128((unsigned)(size_t)&sA[row][col],     src);
  async_copy_b128((unsigned)(size_t)&sA[row][col + 8], src + 8);
}

__global__ __launch_bounds__(256) void out_gemm_kernel(
    const __bf16* __restrict__ A, const float* __restrict__ Wo,
    const float* __restrict__ bo, float* __restrict__ out)
{
  __shared__ __bf16 sA [2][128][32];
  __shared__ __bf16 sBt[2][128][32];

  const int nb = blockIdx.x * 128;
  const int mb = blockIdx.y * 128;
  const int tid  = threadIdx.x;
  const int wave = tid >> 5, lane = tid & 31;
  const int lh   = lane >> 4, ln  = lane & 15;
  const int wm   = (wave >> 1) * 32, wn = (wave & 1) * 64;

  const __bf16* ab = &A[(size_t)mb * D_MODEL];
  const float*  wb = &Wo[nb];

  v8f zero = {};
  v8f acc[2][4];
  #pragma unroll
  for (int mt = 0; mt < 2; mt++)
    #pragma unroll
    for (int nt = 0; nt < 4; nt++) acc[mt][nt] = zero;

  stage_bf16_A_async(sA[0], ab, tid);
  stage_f32_Bt(sBt[0], wb, tid);
  async_wait0();
  __syncthreads();

  for (int k0 = 0; k0 < D_MODEL; k0 += 32) {
    const int cur = (k0 >> 5) & 1;
    if (k0 + 32 < D_MODEL) {
      stage_bf16_A_async(sA[cur ^ 1], ab + (k0 + 32), tid);
      stage_f32_Bt(sBt[cur ^ 1], wb + (size_t)(k0 + 32) * D_MODEL, tid);
    }

    v16bf aF[2], bF[4];
    #pragma unroll
    for (int mt = 0; mt < 2; mt++)
      #pragma unroll
      for (int i = 0; i < 16; i++)
        aF[mt][i] = sA[cur][wm + mt * 16 + ln][afrag_k(lane, i)];
    #pragma unroll
    for (int nt = 0; nt < 4; nt++)
      #pragma unroll
      for (int i = 0; i < 16; i++)
        bF[nt][i] = sBt[cur][wn + nt * 16 + ln][lh * 16 + i];

    #pragma unroll
    for (int mt = 0; mt < 2; mt++)
      #pragma unroll
      for (int nt = 0; nt < 4; nt++)
        acc[mt][nt] = wmma_bf16(aF[mt], bF[nt], acc[mt][nt]);

    async_wait0();
    __syncthreads();
  }

  #pragma unroll
  for (int mt = 0; mt < 2; mt++)
    #pragma unroll
    for (int nt = 0; nt < 4; nt++)
      #pragma unroll
      for (int r = 0; r < 8; r++) {
        const int grow = mb + wm + mt * 16 + r + lh * 8;
        const int gcol = nb + wn + nt * 16 + ln;
        out[(size_t)grow * D_MODEL + gcol] = acc[mt][nt][r] + bo[gcol];
      }
}

// ==========================================================================
extern "C" void kernel_launch(void* const* d_in, const int* in_sizes, int n_in,
                              void* d_out, int out_size, void* d_ws, size_t ws_size,
                              hipStream_t stream) {
  (void)in_sizes; (void)n_in; (void)out_size; (void)ws_size;
  const float* x  = (const float*)d_in[0];
  const float* Wq = (const float*)d_in[1];
  const float* bq = (const float*)d_in[2];
  const float* Wk = (const float*)d_in[3];
  const float* bk = (const float*)d_in[4];
  const float* Wv = (const float*)d_in[5];
  const float* bv = (const float*)d_in[6];
  const float* Wo = (const float*)d_in[7];
  const float* bo = (const float*)d_in[8];
  float* out = (float*)d_out;

  char* ws = (char*)d_ws;
  __bf16* Q  = (__bf16*)(ws);                               // 8 MB
  __bf16* K  = (__bf16*)(ws + (size_t)8  * 1024 * 1024);    // 8 MB
  __bf16* V  = (__bf16*)(ws + (size_t)16 * 1024 * 1024);    // 8 MB
  __bf16* AO = (__bf16*)(ws + (size_t)24 * 1024 * 1024);    // 8 MB

  dim3 blk(256);
  qkv_gemm_kernel<<<dim3(8, 32, 3), blk, 0, stream>>>(x, Wq, bq, Wk, bk, Wv, bv, Q, K, V);
  attn_kernel    <<<dim3(16, 32),   blk, 0, stream>>>(Q, K, V, AO);
  out_gemm_kernel<<<dim3(8, 32),    blk, 0, stream>>>(AO, Wo, bo, out);
}
